// MS_Attention_73787538145800
// MI455X (gfx1250) — matrix operand order repaired and checked by reading, not compiled, not run on hardware
//
#include <hip/hip_runtime.h>
#include <hip/hip_bf16.h>

// ---------------------------------------------------------------------------
// MS attention pipeline for MI455X (gfx1250, wave32)
//  - all matmuls on v_wmma_f32_16x16x32_bf16 (fp32 accumulate)
//  - bf16 intermediate activations (HBM traffic halved vs fp32)
//  - GEMMs software-pipelined: double-buffered LDS, loads overlap WMMA
//  - attention K tiles staged with GLOBAL_LOAD_ASYNC_TO_LDS_B128 (ASYNCcnt)
// ---------------------------------------------------------------------------

typedef __bf16 bf16;
typedef bf16  v16bf __attribute__((ext_vector_type(16)));
typedef bf16  v8bf  __attribute__((ext_vector_type(8)));
typedef float v8f   __attribute__((ext_vector_type(8)));

static constexpr int kB   = 16;    // batch
static constexpr int kNQ  = 1024;  // queries
static constexpr int kNP  = 1024;  // keys (scale 1)
static constexpr int kC   = 512;   // channels
static constexpr int kNUP = 2048;  // upsampled points (scale 2)
static constexpr int kHD  = 64;    // head dim

__device__ __forceinline__ v8f wmma_bf(v16bf a, v16bf b, v8f c) {
  return __builtin_amdgcn_wmma_f32_16x16x32_bf16(false, a, false, b,
                                                 (short)0, c, false, false);
}
__device__ __forceinline__ bf16 f2bf(float f) { return (bf16)f; }

// A fragment (16x32 bf16) from a row-major bf16 row pointer.
// lanes 0-15: K = kb+[0..15]; lanes 16-31: kb+[16..31]
__device__ __forceinline__ v16bf afrag(const bf16* rowp, int half, int kb) {
  v8bf lo = *(const v8bf*)(rowp + kb + half * 16);
  v8bf hi = *(const v8bf*)(rowp + kb + half * 16 + 8);
  v16bf r;
#pragma unroll
  for (int i = 0; i < 8; ++i) { r[i] = lo[i]; r[8 + i] = hi[i]; }
  return r;
}
__device__ __forceinline__ v16bf bfrag(const bf16* p) { return *(const v16bf*)p; }

// Async DMA: 16 bytes per lane, global -> LDS, tracked by ASYNCcnt.
__device__ __forceinline__ void async_ld_b128(unsigned lds_off, const void* gp) {
  asm volatile("global_load_async_to_lds_b128 %0, %1, off"
               :: "v"(lds_off), "v"(gp) : "memory");
}
__device__ __forceinline__ void async_wait0() {
#if __has_builtin(__builtin_amdgcn_s_wait_asynccnt)
  __builtin_amdgcn_s_wait_asynccnt(0);
#else
  asm volatile("s_wait_asynccnt 0" ::: "memory");
#endif
}

// ---------------------------------------------------------------------------
// GEMM:  Y[M,N] = act( X[M,K] @ W[N,K]^T + bias ), fp32 in, fp32 or bf16 out.
// Block 256 threads (8 waves), tile 128x128, K-step 32, wave tile 32x64.
// Double-buffered LDS software pipeline; optional row gather/scatter.
// ---------------------------------------------------------------------------
__global__ __launch_bounds__(256) void gemm_bf16w(
    const float* __restrict__ X, const float* __restrict__ W,
    const float* __restrict__ bias, void* __restrict__ Yv,
    int M, int N, int K, int relu, int out_bf16,
    int in_remap, int in_inner, int in_outer,
    int out_remap, int out_inner, int out_outer)
{
  __shared__ __align__(32) bf16 Xs[2][128][32];
  __shared__ __align__(32) bf16 Ws[2][128][32];

  const int tid  = threadIdx.x;
  const int wave = tid >> 5;
  const int lane = tid & 31;
  const int m    = lane & 15;
  const int half = lane >> 4;
  const int mt   = blockIdx.y * 128;
  const int nt   = blockIdx.x * 128;
  const int wm   = (wave & 3) * 32;
  const int wn   = (wave >> 2) * 64;

  // hoist staging addresses (incl. gather remap) out of the K loop
  const float* xp[4];
  const float* wp[4];
  int srow[4], scol[4];
#pragma unroll
  for (int it = 0; it < 4; ++it) {
    int f   = it * 256 + tid;   // float4 slot: 8 per 32-wide row
    int row = f >> 3;
    int c4  = (f & 7) * 4;
    srow[it] = row; scol[it] = c4;
    int gr = mt + row;
    int sr = in_remap ? ((gr % in_outer) * in_inner + gr / in_outer) : gr;
    xp[it] = X + (size_t)sr * K + c4;
    wp[it] = W + (size_t)(nt + row) * K + c4;
  }

  const v8f vz = {0.f, 0.f, 0.f, 0.f, 0.f, 0.f, 0.f, 0.f};
  v8f acc[2][4];
#pragma unroll
  for (int i = 0; i < 2; ++i)
#pragma unroll
    for (int j = 0; j < 4; ++j) acc[i][j] = vz;

  float4 xv[4], wv[4];

  auto load_regs = [&](int k0) {   // batch all 8 loads -> one wait
#pragma unroll
    for (int it = 0; it < 4; ++it) {
      xv[it] = *(const float4*)(xp[it] + k0);
      wv[it] = *(const float4*)(wp[it] + k0);
    }
    if (k0 + 32 < K) {
      __builtin_prefetch(xp[0] + k0 + 32, 0, 3);
      __builtin_prefetch(wp[0] + k0 + 32, 0, 3);
    }
  };
  auto store_tile = [&](int buf) {
#pragma unroll
    for (int it = 0; it < 4; ++it) {
      int row = srow[it], c4 = scol[it];
      Xs[buf][row][c4 + 0] = f2bf(xv[it].x); Xs[buf][row][c4 + 1] = f2bf(xv[it].y);
      Xs[buf][row][c4 + 2] = f2bf(xv[it].z); Xs[buf][row][c4 + 3] = f2bf(xv[it].w);
      Ws[buf][row][c4 + 0] = f2bf(wv[it].x); Ws[buf][row][c4 + 1] = f2bf(wv[it].y);
      Ws[buf][row][c4 + 2] = f2bf(wv[it].z); Ws[buf][row][c4 + 3] = f2bf(wv[it].w);
    }
  };
  auto compute = [&](int buf) {
    v16bf af0 = afrag(&Xs[buf][wm +  0 + m][0], half, 0);
    v16bf af1 = afrag(&Xs[buf][wm + 16 + m][0], half, 0);
#pragma unroll
    for (int j = 0; j < 4; ++j) {
      v16bf bv = bfrag(&Ws[buf][wn + j * 16 + m][half * 16]);
      acc[0][j] = wmma_bf(af0, bv, acc[0][j]);
      acc[1][j] = wmma_bf(af1, bv, acc[1][j]);
    }
  };

  // software pipeline: loads for tile t+1 in flight while tile t runs on WMMA
  load_regs(0);
  store_tile(0);
  __syncthreads();
  const int nk = K >> 5;
  for (int t = 1; t < nk; ++t) {
    load_regs(t * 32);          // issue next tile's global loads
    compute((t - 1) & 1);       // WMMAs hide the load latency
    store_tile(t & 1);          // convert + stage into alternate buffer
    __syncthreads();
  }
  compute((nk - 1) & 1);

  float* Yf = (float*)Yv;
  bf16*  Yb = (bf16*)Yv;
#pragma unroll
  for (int i = 0; i < 2; ++i) {
#pragma unroll
    for (int j = 0; j < 4; ++j) {
      int gc = nt + wn + j * 16 + m;
      float bv = bias ? bias[gc] : 0.f;
#pragma unroll
      for (int r = 0; r < 8; ++r) {
        int gr = mt + wm + i * 16 + r + 8 * half;
        float v = acc[i][j][r] + bv;
        if (relu) v = fmaxf(v, 0.f);
        int orow = out_remap ? ((gr % out_inner) * out_outer + gr / out_inner) : gr;
        if (out_bf16) Yb[(size_t)orow * N + gc] = f2bf(v);
        else          Yf[(size_t)orow * N + gc] = v;
      }
    }
  }
}

// ---------------------------------------------------------------------------
// 3-NN + normalized inverse-distance weights.
// ---------------------------------------------------------------------------
__global__ __launch_bounds__(256) void nn3_top3(
    const float* __restrict__ xyzup, const float* __restrict__ keypos,
    int* __restrict__ idxo, float* __restrict__ wo)
{
  int t = blockIdx.x * blockDim.x + threadIdx.x;
  if (t >= kB * kNUP) return;
  int b = t / kNUP;
  const float px = xyzup[(size_t)t * 3 + 0];
  const float py = xyzup[(size_t)t * 3 + 1];
  const float pz = xyzup[(size_t)t * 3 + 2];
  const float* kp = keypos + (size_t)b * kNP * 3;
  float d0 = 3e38f, d1 = 3e38f, d2 = 3e38f;
  int   i0 = 0, i1 = 0, i2 = 0;
  for (int p = 0; p < kNP; ++p) {
    float dx = kp[p * 3 + 0] - px;
    float dy = kp[p * 3 + 1] - py;
    float dz = kp[p * 3 + 2] - pz;
    float d  = dx * dx + dy * dy + dz * dz;
    if (d < d0)      { d2 = d1; i2 = i1; d1 = d0; i1 = i0; d0 = d; i0 = p; }
    else if (d < d1) { d2 = d1; i2 = i1; d1 = d;  i1 = p; }
    else if (d < d2) { d2 = d;  i2 = p; }
  }
  float w0 = 1.f / (d0 + 1e-8f);
  float w1 = 1.f / (d1 + 1e-8f);
  float w2 = 1.f / (d2 + 1e-8f);
  float s  = w0 + w1 + w2;
  wo[t * 3 + 0] = w0 / s; wo[t * 3 + 1] = w1 / s; wo[t * 3 + 2] = w2 / s;
  idxo[t * 3 + 0] = i0;   idxo[t * 3 + 1] = i1;   idxo[t * 3 + 2] = i2;
}

// Weighted gather of key features (key is (Np,B,C) fp32) -> interp (B*NUP, C)
__global__ __launch_bounds__(128) void interp_gather(
    const float* __restrict__ key, const int* __restrict__ idxo,
    const float* __restrict__ wo, float* __restrict__ outp)
{
  int t = blockIdx.x;  // b*NUP + i
  int b = t / kNUP;
  int i0 = idxo[t * 3 + 0], i1 = idxo[t * 3 + 1], i2 = idxo[t * 3 + 2];
  float w0 = wo[t * 3 + 0], w1 = wo[t * 3 + 1], w2 = wo[t * 3 + 2];
  const float* f0 = key + ((size_t)i0 * kB + b) * kC;
  const float* f1 = key + ((size_t)i1 * kB + b) * kC;
  const float* f2 = key + ((size_t)i2 * kB + b) * kC;
  float* o = outp + (size_t)t * kC;
  for (int c = threadIdx.x; c < kC; c += 128)
    o[c] = w0 * f0[c] + w1 * f1[c] + w2 * f2[c];
}

// ---------------------------------------------------------------------------
// Flash attention. Q/K/V are bf16 (from projection GEMMs).
// Block = (64 q-rows, head, batch); 4 waves x 16 q-rows.
// K tile: async DMA global->LDS (raw bf16 rows). V tile: register transpose.
// ---------------------------------------------------------------------------
__global__ __launch_bounds__(128) void attn_flash(
    const bf16* __restrict__ qp, const bf16* __restrict__ kv,
    float* __restrict__ xcat, int Nkv, int qh_off, int xcol_off)
{
  __shared__ __align__(32) bf16 Kt[64][64];      // [key][hd]
  __shared__ __align__(32) bf16 Vt[64][64];      // [hd][key] (transposed)
  __shared__ __align__(32) bf16 Pw[4][16][64];   // per-wave P relayout buffer

  const int tid  = threadIdx.x;
  const int wave = tid >> 5;
  const int lane = tid & 31;
  const int m    = lane & 15;
  const int half = lane >> 4;
  const int qt = blockIdx.x, h = blockIdx.y, b = blockIdx.z;
  const int qrow0 = qt * 64 + wave * 16;
  const float scale = 0.125f;  // hd^-0.5, hd = 64
  const unsigned kt_base = (unsigned)(uintptr_t)(&Kt[0][0]);

  // Q fragments: 16 rows x 64 hd as two 16x32 A fragments (bf16, direct loads)
  const bf16* qrow = qp + ((size_t)(b * kNQ + qrow0 + m)) * kC + (qh_off + h) * kHD;
  v16bf qa[2];
#pragma unroll
  for (int kb = 0; kb < 2; ++kb) {
    v8bf lo = *(const v8bf*)(qrow + kb * 32 + half * 16);
    v8bf hi = *(const v8bf*)(qrow + kb * 32 + half * 16 + 8);
    v16bf u;
#pragma unroll
    for (int i = 0; i < 8; ++i) { u[i] = lo[i]; u[8 + i] = hi[i]; }
    qa[kb] = u;
  }

  // hoisted per-thread staging descriptors
  const bf16* kseg[4];
  unsigned    ksegl[4];
#pragma unroll
  for (int i = 0; i < 4; ++i) {
    int s = i * 128 + tid;              // 512 segments of 16B per K tile
    int keyi = s >> 3, c8 = (s & 7) * 8;
    kseg[i]  = kv + ((size_t)(b * Nkv + keyi)) * kC + h * kHD + c8;
    ksegl[i] = kt_base + (unsigned)(keyi * 128 + (s & 7) * 16);
  }
  const int vkey = tid >> 1, vdh = (tid & 1) * 32;
  const bf16* vseg = kv + ((size_t)(b * Nkv + vkey)) * kC + 256 + h * kHD + vdh;

  float mprev[8], lrun[8];
#pragma unroll
  for (int r = 0; r < 8; ++r) { mprev[r] = -3.0e30f; lrun[r] = 0.f; }
  const v8f vz = {0.f, 0.f, 0.f, 0.f, 0.f, 0.f, 0.f, 0.f};
  v8f oacc[4];
#pragma unroll
  for (int j = 0; j < 4; ++j) oacc[j] = vz;

  for (int p0 = 0; p0 < Nkv; p0 += 64) {
    __syncthreads();  // previous tile fully consumed
    const size_t tileoff = (size_t)p0 * kC;

    // K tile: 64 keys x 64 bf16 = 8KB via async DMA (16B/lane segments)
#pragma unroll
    for (int i = 0; i < 4; ++i) async_ld_b128(ksegl[i], kseg[i] + tileoff);

    // V tile: batch loads, then transpose-store into LDS
    v8bf vv[4];
#pragma unroll
    for (int i = 0; i < 4; ++i) vv[i] = *(const v8bf*)(vseg + tileoff + i * 8);
#pragma unroll
    for (int i = 0; i < 4; ++i)
#pragma unroll
      for (int e = 0; e < 8; ++e) Vt[vdh + i * 8 + e][vkey] = vv[i][e];

    async_wait0();    // wave's async DMA landed in LDS
    __syncthreads();  // all waves' K/V staged

    // S = Q K^T (16 x 64) as 4 fragments
    v8f s[4];
#pragma unroll
    for (int c = 0; c < 4; ++c) {
      v8f a = vz;
      a = wmma_bf(qa[0], bfrag(&Kt[c * 16 + m][0 + half * 16]), a);
      a = wmma_bf(qa[1], bfrag(&Kt[c * 16 + m][32 + half * 16]), a);
      s[c] = a;
    }

    // online softmax (rows striped across 16-lane groups -> shfl_xor reduce)
#pragma unroll
    for (int r = 0; r < 8; ++r) {
      float rm = fmaxf(fmaxf(s[0][r], s[1][r]), fmaxf(s[2][r], s[3][r])) * scale;
      rm = fmaxf(rm, __shfl_xor(rm, 1, 32));
      rm = fmaxf(rm, __shfl_xor(rm, 2, 32));
      rm = fmaxf(rm, __shfl_xor(rm, 4, 32));
      rm = fmaxf(rm, __shfl_xor(rm, 8, 32));
      float nm    = fmaxf(mprev[r], rm);
      float alpha = __expf(mprev[r] - nm);
      mprev[r]    = nm;
      float rs = 0.f;
#pragma unroll
      for (int c = 0; c < 4; ++c) {
        float p = __expf(s[c][r] * scale - nm);
        Pw[wave][r + 8 * half][c * 16 + m] = f2bf(p);  // C-layout -> row-major
        rs += p;
      }
      rs += __shfl_xor(rs, 1, 32);
      rs += __shfl_xor(rs, 2, 32);
      rs += __shfl_xor(rs, 4, 32);
      rs += __shfl_xor(rs, 8, 32);
      lrun[r] = lrun[r] * alpha + rs;
#pragma unroll
      for (int j = 0; j < 4; ++j) oacc[j][r] = oacc[j][r] * alpha;
    }

    // O += P V  (per-wave LDS round trip gives the A-layout for P)
#pragma unroll
    for (int kc = 0; kc < 2; ++kc) {
      v16bf pa = afrag(&Pw[wave][m][0], half, kc * 32);
#pragma unroll
      for (int j = 0; j < 4; ++j)
        oacc[j] = wmma_bf(pa, bfrag(&Vt[j * 16 + m][kc * 32 + half * 16]), oacc[j]);
    }
  }

  // normalize and write into concatenated fp32 (B, Nq, 512) buffer
#pragma unroll
  for (int r = 0; r < 8; ++r) {
    float inv = 1.f / lrun[r];
    int row = qrow0 + r + 8 * half;
    float* op = xcat + ((size_t)(b * kNQ + row)) * kC + xcol_off + h * kHD;
#pragma unroll
    for (int j = 0; j < 4; ++j) op[j * 16 + m] = oacc[j][r] * inv;
  }
}

// ---------------------------------------------------------------------------
// Host-side orchestration
// ---------------------------------------------------------------------------
extern "C" void kernel_launch(void* const* d_in, const int* in_sizes, int n_in,
                              void* d_out, int out_size, void* d_ws, size_t ws_size,
                              hipStream_t stream)
{
  (void)in_sizes; (void)n_in; (void)out_size; (void)ws_size;
  const float* query  = (const float*)d_in[0];   // (Nq, B, C)
  const float* key    = (const float*)d_in[1];   // (Np, B, C)
  const float* keypos = (const float*)d_in[2];   // (B, Np, 3)
  // d_in[3] xyz_all unused (matches reference)
  const float* xyzup  = (const float*)d_in[4];   // (B, n_up, 3)
  const float* Wq     = (const float*)d_in[5];
  const float* Wkv1   = (const float*)d_in[6];
  const float* Wkv2   = (const float*)d_in[7];
  const float* fpw    = (const float*)d_in[8];
  const float* fpb    = (const float*)d_in[9];
  const float* projw  = (const float*)d_in[10];
  const float* projb  = (const float*)d_in[11];
  float* out = (float*)d_out;

  char* ws = (char*)d_ws;
  size_t off = 0;
  auto alloc = [&](size_t bytes) -> void* {
    void* p = ws + off;
    off += (bytes + 255) & ~(size_t)255;
    return p;
  };
  bf16*  qproj  = (bf16*) alloc((size_t)kB * kNQ  * kC * 2);
  bf16*  kv1    = (bf16*) alloc((size_t)kB * kNP  * kC * 2);
  bf16*  kv2    = (bf16*) alloc((size_t)kB * kNUP * kC * 2);
  float* interp = (float*)alloc((size_t)kB * kNUP * kC * 4);
  float* key2   = (float*)alloc((size_t)kB * kNUP * kC * 4);
  float* xcat   = (float*)alloc((size_t)kB * kNQ  * kC * 4);
  int*   nnidx  = (int*)  alloc((size_t)kB * kNUP * 3 * 4);
  float* nnw    = (float*)alloc((size_t)kB * kNUP * 3 * 4);

  const int M1 = kB * kNQ;   // 16384 rows
  const int M2 = kB * kNUP;  // 32768 rows
  dim3 blk(256);

  // q projection -> bf16 (flat rows == torch's (B, Nq, H, hd) reinterpretation)
  gemm_bf16w<<<dim3(kC / 128, M1 / 128), blk, 0, stream>>>(
      query, Wq, nullptr, qproj, M1, kC, kC, 0, 1, 0, 0, 0, 0, 0, 0);
  // kv1 projection -> bf16
  gemm_bf16w<<<dim3(kC / 128, M1 / 128), blk, 0, stream>>>(
      key, Wkv1, nullptr, kv1, M1, kC, kC, 0, 1, 0, 0, 0, 0, 0, 0);
  // 3-NN + inverse-distance weights
  nn3_top3<<<(kB * kNUP + 255) / 256, 256, 0, stream>>>(xyzup, keypos, nnidx, nnw);
  // weighted feature gather
  interp_gather<<<kB * kNUP, 128, 0, stream>>>(key, nnidx, nnw, interp);
  // FP pointwise MLP + ReLU -> key2 (B, n_up, C) fp32
  gemm_bf16w<<<dim3(kC / 128, M2 / 128), blk, 0, stream>>>(
      interp, fpw, fpb, key2, M2, kC, kC, 1, 0, 0, 0, 0, 0, 0, 0);
  // kv2 projection -> bf16, with (B,n)->(n,B) input row gather (transpose fused)
  gemm_bf16w<<<dim3(kC / 128, M2 / 128), blk, 0, stream>>>(
      key2, Wkv2, nullptr, kv2, M2, kC, kC, 0, 1, 1, kNUP, kB, 0, 0, 0);
  // attention scale 1: heads 0..3, 1024 keys -> xcat cols [0,256)
  attn_flash<<<dim3(kNQ / 64, 4, kB), dim3(128), 0, stream>>>(
      qproj, kv1, xcat, kNP, 0, 0);
  // attention scale 2: heads 4..7, 2048 keys -> xcat cols [256,512)
  attn_flash<<<dim3(kNQ / 64, 4, kB), dim3(128), 0, stream>>>(
      qproj, kv2, xcat, kNUP, 4, 256);
  // final projection (fp32) with (B,Nq)->(Nq,B) output row scatter
  gemm_bf16w<<<dim3(kC / 128, M1 / 128), blk, 0, stream>>>(
      xcat, projw, projb, out, M1, kC, kC, 0, 0, 0, 0, 0, 1, kNQ, kB);
}
